// MaskedBlockViT_26603027432214
// MI455X (gfx1250) — matrix operand order repaired and checked by reading, not compiled, not run on hardware
//
#include <hip/hip_runtime.h>
#include <hip/hip_bf16.h>
#include <math.h>

// ---------------------------------------------------------------------------
// MaskedBlockViT for gfx1250 (MI455X, wave32, WMMA + Tensor Data Mover).
// GEMMs run on v_wmma_f32_16x16x32_f16 (f16 in, f32 accumulate); the A
// operand is staged through LDS with TDM (tensor_load_to_lds, TENSORcnt).
// ---------------------------------------------------------------------------

typedef _Float16 half_t;
typedef _Float16 v16h __attribute__((ext_vector_type(16)));
typedef float    v8f  __attribute__((ext_vector_type(8)));

#define DEV static __device__ __forceinline__

DEV float geluf(float x) { return 0.5f * x * (1.0f + erff(x * 0.70710678118654752f)); }
DEV int imin(int a, int b) { return a < b ? a : b; }

// ------------------------- conversion kernels ------------------------------
__global__ void k_f32_to_f16(const float* __restrict__ in, half_t* __restrict__ out, long n) {
  long i = blockIdx.x * (long)blockDim.x + threadIdx.x;
  if (i < n) out[i] = (half_t)in[i];
}

// W is K x N (row-major f32) -> WT is N x K (row-major f16)
__global__ void k_transpose_f16(const float* __restrict__ W, half_t* __restrict__ WT, int K, int N) {
  long i = blockIdx.x * (long)blockDim.x + threadIdx.x;
  if (i >= (long)K * N) return;
  int k = (int)(i / N), n = (int)(i % N);
  WT[(long)n * K + k] = (half_t)W[i];
}

__global__ void k_zero_f32(float* __restrict__ p, long n) {
  long i = blockIdx.x * (long)blockDim.x + threadIdx.x;
  if (i < n) p[i] = 0.0f;
}

// ------------------------- WMMA GEMM ---------------------------------------
// C[M,N] = act( A[M,K] (f16, row-major) * B[K,N] + bias ), B given as
// BT[N,K] (f16, row-major).  act: 0=none 1=gelu 2=relu.  K % 64 == 0,
// N % 64 == 0.  Grid: (N/64, ceil(M/32)), block 256 (8 wave32 waves; each
// wave owns one 16x16 C tile of a 32x64 block tile).
//
// Fragment layouts used here (wave32; A/B use contiguous k, not the ISA table order):
//   A 16x32 f16 : lane L holds row M=L%16; half h -> K=16*(L/16)+h
//   B 32x16 f16 : lane L holds col N=L%16; half h -> K=16*(L/16)+h
//   C/D 16x16 f32: lane L holds col N=L%16; VGPR i -> row M=i+8*(L/16)

#if __has_builtin(__builtin_amdgcn_tensor_load_to_lds)
#define HAVE_TDM 1
#endif

#ifdef HAVE_TDM

#define KT 64           // K-halves per TDM stage
#define APAD 72         // LDS row stride in halves (64 data + 8 pad, via D# pad)

typedef unsigned int u32x4 __attribute__((ext_vector_type(4)));
typedef int          i32x4 __attribute__((ext_vector_type(4)));
typedef int          i32x8 __attribute__((ext_vector_type(8)));

// Issue a 2D TDM load of a [rows_valid x KT]-half tile (row stride K halves)
// from global into LDS at lds_off, with 4-dword padding after each 32-dword
// row (LDS row stride = APAD halves).  Rows >= rows_valid are zero-filled by
// the TDM OOB rule.  D# packing per CDNA5 ISA 8.3/8.4.
DEV void tdm_load_a(const half_t* gsrc, unsigned lds_off, int rows_valid, int K) {
  unsigned long long ga = (unsigned long long)gsrc;
  u32x4 g0 = {};
  g0[0] = 1u;                                             // count=1, user desc
  g0[1] = lds_off;                                        // lds_addr (bytes)
  g0[2] = (unsigned)(ga & 0xffffffffull);                 // global_addr[31:0]
  g0[3] = (unsigned)((ga >> 32) & 0x1ffffffull)           // global_addr[56:32]
          | 0x80000000u;                                  // type=2 ("image")
  i32x8 g1 = {};
  unsigned td0 = (unsigned)KT;                            // tensor_dim0
  unsigned td1 = (unsigned)rows_valid;                    // tensor_dim1 (OOB rows -> 0)
  g1[0] = (int)(0x10000u                                  // data_size=1 (2 bytes)
                | (1u << 20)                              // pad_enable
                | (4u << 22)                              // pad_interval: 32 dwords
                | (3u << 25));                            // pad_amount: 4 dwords
  g1[1] = (int)((td0 & 0xffffu) << 16);                   // tensor_dim0[15:0]
  g1[2] = (int)((td0 >> 16) | ((td1 & 0xffffu) << 16));   // td0[31:16] | td1[15:0]
  g1[3] = (int)((td1 >> 16) | ((unsigned)KT << 16));      // td1[31:16] | tile_dim0
  g1[4] = 32;                                             // tile_dim1=32, tile_dim2=0
  g1[5] = K;                                              // tensor_dim0_stride[31:0]
  g1[6] = 0;
  g1[7] = 0;
  i32x4 z4 = {};
#if defined(__clang_major__) && (__clang_major__ >= 23)
  i32x8 z8 = {};
  __builtin_amdgcn_tensor_load_to_lds(g0, g1, z4, z4, z8, 0);
#else
  __builtin_amdgcn_tensor_load_to_lds(g0, g1, z4, z4, 0);
#endif
}

__global__ void __launch_bounds__(256) k_wmma_gemm(
    const half_t* __restrict__ A, const half_t* __restrict__ BT,
    const float* __restrict__ bias, float* __restrict__ C,
    int M, int N, int K, int act) {
  __shared__ half_t sA[2][32][APAD];   // double-buffered A stage, 9216 B
  const int lane = threadIdx.x & 31;
  const int wave = threadIdx.x >> 5;
  const int wm = wave >> 2;   // 0..1
  const int wn = wave & 3;    // 0..3
  const int l16 = lane & 15;
  const int lg  = lane >> 4;  // 0/1
  const int row0 = blockIdx.y * 32 + wm * 16;
  const int col0 = blockIdx.x * 64 + wn * 16;
  const int rows_valid = imin(32, M - blockIdx.y * 32);
  const half_t* gA = A + (long)(blockIdx.y * 32) * K;     // block A tile base
  const unsigned lds0 = (unsigned)(unsigned long long)(&sA[0][0][0]);
  const unsigned lds1 = (unsigned)(unsigned long long)(&sA[1][0][0]);

  int bcol = col0 + l16; if (bcol > N - 1) bcol = N - 1;  // EXEC must stay full
  const half_t* bp = BT + (long)bcol * K + 16 * lg;

  if (wave == 0) {                    // TDM issue is per-wave; wave 0 drives it
    tdm_load_a(gA, lds0, rows_valid, K);
    __builtin_amdgcn_s_wait_tensorcnt(0);
  }
  __syncthreads();

  v8f acc = {};
  const int nstage = K >> 6;          // K % 64 == 0
  for (int s = 0; s < nstage; ++s) {
    if (s + 1 < nstage && wave == 0)
      tdm_load_a(gA + (long)(s + 1) * KT, ((s + 1) & 1) ? lds1 : lds0, rows_valid, K);
    const half_t* arow = &sA[s & 1][wm * 16 + l16][0];
#pragma unroll
    for (int ks = 0; ks < KT; ks += 32) {
      v16h a, b;
#pragma unroll
      for (int h = 0; h < 8; ++h) {
        a[h]     = arow[ks + 16 * lg + h];
        a[h + 8] = arow[ks + 16 * lg + 8 + h];
      }
#pragma unroll
      for (int h = 0; h < 16; ++h) b[h] = bp[h];
      __builtin_prefetch(bp + KT, 0, 0);   // global_prefetch_b8, next B stage
      acc = __builtin_amdgcn_wmma_f32_16x16x32_f16(false, a, false, b,
                                                   (short)0, acc, false, false);
      bp += 32;
    }
    if (wave == 0) __builtin_amdgcn_s_wait_tensorcnt(0);
    __syncthreads();
  }
#pragma unroll
  for (int i = 0; i < 8; ++i) {
    int r = row0 + 8 * lg + i;
    int c = col0 + l16;
    if (r < M && c < N) {
      float v = acc[i] + (bias ? bias[c] : 0.0f);
      if (act == 1) v = geluf(v); else if (act == 2) v = fmaxf(v, 0.0f);
      C[(long)r * N + c] = v;
    }
  }
}

#else  // !HAVE_TDM: direct-global fragment path

__global__ void __launch_bounds__(256) k_wmma_gemm(
    const half_t* __restrict__ A, const half_t* __restrict__ BT,
    const float* __restrict__ bias, float* __restrict__ C,
    int M, int N, int K, int act) {
  const int lane = threadIdx.x & 31;
  const int wave = threadIdx.x >> 5;
  const int wm = wave >> 2;
  const int wn = wave & 3;
  const int l16 = lane & 15;
  const int lg  = lane >> 4;
  const int row0 = blockIdx.y * 32 + wm * 16;
  const int col0 = blockIdx.x * 64 + wn * 16;
  int arow = row0 + l16; if (arow > M - 1) arow = M - 1;
  int bcol = col0 + l16; if (bcol > N - 1) bcol = N - 1;
  const half_t* ap = A  + (long)arow * K + 16 * lg;
  const half_t* bp = BT + (long)bcol * K + 16 * lg;
  v8f acc = {};
  for (int kb = 0; kb < K; kb += 32) {
    v16h a, b;
#pragma unroll
    for (int h = 0; h < 8; ++h) { a[h] = ap[h]; a[h + 8] = ap[h + 8]; }
#pragma unroll
    for (int h = 0; h < 16; ++h) b[h] = bp[h];
    __builtin_prefetch(ap + 32, 0, 0);
    __builtin_prefetch(bp + 32, 0, 0);
    acc = __builtin_amdgcn_wmma_f32_16x16x32_f16(false, a, false, b,
                                                 (short)0, acc, false, false);
    ap += 32; bp += 32;
  }
#pragma unroll
  for (int i = 0; i < 8; ++i) {
    int r = row0 + 8 * lg + i;
    int c = col0 + l16;
    if (r < M && c < N) {
      float v = acc[i] + (bias ? bias[c] : 0.0f);
      if (act == 1) v = geluf(v); else if (act == 2) v = fmaxf(v, 0.0f);
      C[(long)r * N + c] = v;
    }
  }
}

#endif  // HAVE_TDM

// ------------------------- direct 3x3 conv (pad 1) -------------------------
__global__ void k_conv3x3(const float* __restrict__ x, const float* __restrict__ w,
                          const float* __restrict__ bias, float* __restrict__ y,
                          int BN, int Cin, int Hin, int Win,
                          int Cout, int Hout, int Wout, int stride, int act) {
  long idx = blockIdx.x * (long)blockDim.x + threadIdx.x;
  long total = (long)BN * Cout * Hout * Wout;
  if (idx >= total) return;
  int ox = (int)(idx % Wout); long t = idx / Wout;
  int oy = (int)(t % Hout); t /= Hout;
  int co = (int)(t % Cout); int n = (int)(t / Cout);
  float acc = bias[co];
  const float* xn = x + (long)n * Cin * Hin * Win;
  const float* wc = w + (long)co * Cin * 9;
  for (int ci = 0; ci < Cin; ++ci) {
    const float* xc = xn + (long)ci * Hin * Win;
    const float* wk = wc + ci * 9;
#pragma unroll
    for (int ky = 0; ky < 3; ++ky) {
      int iy = oy * stride + ky - 1;
      if (iy < 0 || iy >= Hin) continue;
#pragma unroll
      for (int kx = 0; kx < 3; ++kx) {
        int ix = ox * stride + kx - 1;
        if (ix < 0 || ix >= Win) continue;
        acc += xc[(long)iy * Win + ix] * wk[ky * 3 + kx];
      }
    }
  }
  y[idx] = act ? geluf(acc) : acc;
}

// -------- transposed 3x3 conv: lhs_dilation=2, padding ((1,2),(1,2)) -------
// Output is 2*Hin x 2*Win.
__global__ void k_convT3x3(const float* __restrict__ x, const float* __restrict__ w,
                           const float* __restrict__ bias, float* __restrict__ y,
                           int BN, int Cin, int Hin, int Win, int Cout, int act) {
  int Hout = 2 * Hin, Wout = 2 * Win;
  long idx = blockIdx.x * (long)blockDim.x + threadIdx.x;
  long total = (long)BN * Cout * Hout * Wout;
  if (idx >= total) return;
  int ox = (int)(idx % Wout); long t = idx / Wout;
  int oy = (int)(t % Hout); t /= Hout;
  int co = (int)(t % Cout); int n = (int)(t / Cout);
  float acc = bias[co];
  const float* xn = x + (long)n * Cin * Hin * Win;
  const float* wc = w + (long)co * Cin * 9;
  for (int ci = 0; ci < Cin; ++ci) {
    const float* xc = xn + (long)ci * Hin * Win;
    const float* wk = wc + ci * 9;
#pragma unroll
    for (int ky = 0; ky < 3; ++ky) {
      int p = oy + ky - 1;
      if (p < 0 || (p & 1) || (p >> 1) >= Hin) continue;  // dilated coord must be even
#pragma unroll
      for (int kx = 0; kx < 3; ++kx) {
        int q = ox + kx - 1;
        if (q < 0 || (q & 1) || (q >> 1) >= Win) continue;
        acc += xc[(long)(p >> 1) * Win + (q >> 1)] * wk[ky * 3 + kx];
      }
    }
  }
  y[idx] = act ? geluf(acc) : acc;
}

// ------------------------- masking / shuffle -------------------------------
// Stable argsort by rank (O(N^2), N=1024, one block).
__global__ void k_argsort(const float* __restrict__ noise, int* __restrict__ ids, int n) {
  for (int i = threadIdx.x; i < n; i += blockDim.x) {
    float vi = noise[i];
    int r = 0;
    for (int j = 0; j < n; ++j) {
      float vj = noise[j];
      r += (vj < vi) || (vj == vi && j < i);
    }
    ids[r] = i;
  }
}

__global__ void k_pos_add(float* __restrict__ xemb, const float* __restrict__ row_e,
                          const float* __restrict__ col_e) {
  int i = blockIdx.x * blockDim.x + threadIdx.x;  // 1024*128
  if (i >= 1024 * 128) return;
  int n = i >> 7, e = i & 127;
  xemb[i] += row_e[(n >> 5) * 128 + e] + col_e[(n & 31) * 128 + e];
}

__global__ void k_gather(const float* __restrict__ xemb, const int* __restrict__ ids,
                         float* __restrict__ hseq, int L) {
  long i = blockIdx.x * (long)blockDim.x + threadIdx.x;
  if (i >= (long)L * 128) return;
  int l = (int)(i >> 7), e = (int)(i & 127);
  hseq[i] = xemb[(long)ids[l] * 128 + e];
}

__global__ void k_scatter(const float* __restrict__ hseq, const int* __restrict__ ids,
                          float* __restrict__ dec, int L) {
  long i = blockIdx.x * (long)blockDim.x + threadIdx.x;
  if (i >= (long)L * 128) return;
  int l = (int)(i >> 7), e = (int)(i & 127);
  dec[(long)ids[l] * 128 + e] = hseq[i];
}

// ------------------------- relative-position buckets -----------------------
DEV int bucket_of(int rel) {   // NB=32, me=16, MAXD=128 (matches reference incl. clip)
  int neg = rel < 0 ? 1 : 0;
  int a = rel < 0 ? -rel : rel;
  if (a > 127) a = 127;
  float safe = (float)(a > 16 ? a : 16);
  int logb = 16 + (int)(logf(safe * 0.0625f) * (16.0f / logf(8.0f)));
  int b = neg * 32 + (a < 16 ? a : logb);
  return b < 0 ? 0 : (b > 31 ? 31 : b);
}

__global__ void k_buckets(int* __restrict__ bk, int L) {
  long i = blockIdx.x * (long)blockDim.x + threadIdx.x;
  if (i >= (long)L * L) return;
  int q = (int)(i / L), k = (int)(i % L);
  int rb = bucket_of((q >> 5) - (k >> 5));       // rows of sequence index
  int cb = bucket_of((q & 31) - (k & 31));       // cols of sequence index
  bk[i] = rb | (cb << 8);
}

// ------------------------- attention (HD=16, 8 heads) ----------------------
// One block per (query q, head h). qkv is [L,384]; q at +0, k at +128, v at +256.
__global__ void __launch_bounds__(256) k_attn(
    const float* __restrict__ qkv, const int* __restrict__ buckets,
    const float* __restrict__ tab,  // [32,8] for this layer
    float* __restrict__ o, int L) {
  int q = blockIdx.x, h = blockIdx.y;
  __shared__ float sc[768];
  __shared__ float red[256];
  __shared__ float osum[16];
  const float scale = 0.25f;  // HD^-0.5
  const float* qv = qkv + (long)q * 384 + h * 16;
  float qreg[16];
#pragma unroll
  for (int d = 0; d < 16; ++d) qreg[d] = qv[d];
  float lmax = -1e30f;
  for (int j = threadIdx.x; j < L; j += blockDim.x) {
    const float* kv = qkv + (long)j * 384 + 128 + h * 16;
    float s = 0.0f;
#pragma unroll
    for (int d = 0; d < 16; ++d) s += qreg[d] * kv[d];
    int bk = buckets[(long)q * L + j];
    s = s * scale + tab[(bk & 0xff) * 8 + h] + tab[((bk >> 8) & 0xff) * 8 + h];
    sc[j] = s;
    lmax = fmaxf(lmax, s);
  }
  red[threadIdx.x] = lmax; __syncthreads();
  for (int st = 128; st > 0; st >>= 1) {
    if (threadIdx.x < st) red[threadIdx.x] = fmaxf(red[threadIdx.x], red[threadIdx.x + st]);
    __syncthreads();
  }
  float m = red[0]; __syncthreads();
  float lsum = 0.0f;
  for (int j = threadIdx.x; j < L; j += blockDim.x) {
    float e = __expf(sc[j] - m);
    sc[j] = e;
    lsum += e;
  }
  red[threadIdx.x] = lsum; __syncthreads();
  for (int st = 128; st > 0; st >>= 1) {
    if (threadIdx.x < st) red[threadIdx.x] += red[threadIdx.x + st];
    __syncthreads();
  }
  float inv = 1.0f / red[0];
  if (threadIdx.x < 16) osum[threadIdx.x] = 0.0f;
  __syncthreads();
  float acc[16];
#pragma unroll
  for (int d = 0; d < 16; ++d) acc[d] = 0.0f;
  for (int j = threadIdx.x; j < L; j += blockDim.x) {
    float p = sc[j];
    const float* vv = qkv + (long)j * 384 + 256 + h * 16;
#pragma unroll
    for (int d = 0; d < 16; ++d) acc[d] += p * vv[d];
  }
#pragma unroll
  for (int d = 0; d < 16; ++d) atomicAdd(&osum[d], acc[d]);
  __syncthreads();
  if (threadIdx.x < 16) o[(long)q * 128 + h * 16 + threadIdx.x] = osum[threadIdx.x] * inv;
}

// ------------------------- residual + layernorm (EMB=128) ------------------
__global__ void k_add_ln(float* __restrict__ hseq, const float* __restrict__ delta,
                         const float* __restrict__ g, const float* __restrict__ b) {
  int row = blockIdx.x, e = threadIdx.x;  // blockDim = 128
  __shared__ float s[128];
  long i = (long)row * 128 + e;
  float v = hseq[i] + delta[i];
  s[e] = v; __syncthreads();
  for (int st = 64; st > 0; st >>= 1) { if (e < st) s[e] += s[e + st]; __syncthreads(); }
  float mean = s[0] * (1.0f / 128.0f);
  __syncthreads();
  float d = v - mean;
  s[e] = d * d; __syncthreads();
  for (int st = 64; st > 0; st >>= 1) { if (e < st) s[e] += s[e + st]; __syncthreads(); }
  float var = s[0] * (1.0f / 128.0f);
  hseq[i] = d * rsqrtf(var + 1e-5f) * g[e] + b[e];
}

// ------------------------- output assembly ---------------------------------
// p is (BN,2,52,52) -> out (B,N,52,52,2)
__global__ void k_out_transpose(const float* __restrict__ p, float* __restrict__ out) {
  long i = blockIdx.x * (long)blockDim.x + threadIdx.x;
  long total = 1024L * 2 * 52 * 52;
  if (i >= total) return;
  int xx = (int)(i % 52); long t = i / 52;
  int yy = (int)(t % 52); t /= 52;
  int c = (int)(t % 2); int n = (int)(t / 2);
  out[(((long)n * 52 + yy) * 52 + xx) * 2 + c] = p[i];
}

__global__ void k_out_mask(const int* __restrict__ ids, float* __restrict__ out) {
  int i = blockIdx.x * blockDim.x + threadIdx.x;
  if (i < 308) out[i] = (float)ids[716 + i];   // ids_mask = ids_shuffle[716:]
}

// ---------------------------------------------------------------------------
static inline void* wsalloc(char* base, size_t& off, size_t bytes) {
  void* p = base + off;
  off += (bytes + 255) & ~(size_t)255;
  return p;
}

#define NBLK(n) ((unsigned)((((long)(n)) + 255) / 256))

extern "C" void kernel_launch(void* const* d_in, const int* in_sizes, int n_in,
                              void* d_out, int out_size, void* d_ws, size_t ws_size,
                              hipStream_t stream) {
  (void)in_sizes; (void)n_in; (void)out_size; (void)ws_size;
  const float* x_blocks  = (const float*)d_in[0];
  const float* noise     = (const float*)d_in[1];
  const float* enc_c1_w  = (const float*)d_in[2];
  const float* enc_c1_b  = (const float*)d_in[3];
  const float* enc_c2_w  = (const float*)d_in[4];
  const float* enc_c2_b  = (const float*)d_in[5];
  const float* enc_c3_w  = (const float*)d_in[6];
  const float* enc_c3_b  = (const float*)d_in[7];
  const float* enc_lin_w = (const float*)d_in[8];
  const float* enc_lin_b = (const float*)d_in[9];
  const float* row_embed = (const float*)d_in[10];
  const float* col_embed = (const float*)d_in[11];
  const float* in_proj_w = (const float*)d_in[12];
  const float* in_proj_b = (const float*)d_in[13];
  const float* out_proj_w= (const float*)d_in[14];
  const float* out_proj_b= (const float*)d_in[15];
  const float* rel_tab   = (const float*)d_in[16];
  const float* ln1_g     = (const float*)d_in[17];
  const float* ln1_b     = (const float*)d_in[18];
  const float* ln2_g     = (const float*)d_in[19];
  const float* ln2_b     = (const float*)d_in[20];
  const float* ff1_w     = (const float*)d_in[21];
  const float* ff1_b     = (const float*)d_in[22];
  const float* ff2_w     = (const float*)d_in[23];
  const float* ff2_b     = (const float*)d_in[24];
  const float* dec_lin_w = (const float*)d_in[25];
  const float* dec_lin_b = (const float*)d_in[26];
  const float* dt1_w     = (const float*)d_in[27];
  const float* dt1_b     = (const float*)d_in[28];
  const float* dt2_w     = (const float*)d_in[29];
  const float* dt2_b     = (const float*)d_in[30];
  const float* dc_w      = (const float*)d_in[31];
  const float* dc_b      = (const float*)d_in[32];

  const int L = 716;         // LKEEP
  const long H1 = 1024L * 32 * 50 * 50;    // 81,920,000
  const long H2 = 1024L * 64 * 25 * 25;    // 40,960,000
  const long H3 = 1024L * 64 * 13 * 13;    // 11,075,584 (== 1024*10816)
  const long U1 = 1024L * 64 * 26 * 26;    // 44,302,336
  const long U2 = 1024L * 32 * 52 * 52;    // 88,604,672
  const long PP = 1024L * 2 * 52 * 52;     //  5,537,792

  char* ws = (char*)d_ws;
  size_t off = 0;
  // Large regions with lifetime reuse.
  float*  bigA  = (float*) wsalloc(ws, off, sizeof(float) * U2);   // h1, later u2
  float*  bigB  = (float*) wsalloc(ws, off, sizeof(float) * U1);   // h2, later u1
  float*  bigC  = (float*) wsalloc(ws, off, sizeof(float) * H3);   // h3, later lin, later p
  half_t* bigH  = (half_t*)wsalloc(ws, off, sizeof(half_t) * H3);  // h3 f16
  half_t* wbigH = (half_t*)wsalloc(ws, off, sizeof(half_t) * 10816L * 128); // enc_lin^T then dec_lin^T
  float*  xemb  = (float*) wsalloc(ws, off, sizeof(float) * 1024 * 128);
  int*    ids   = (int*)   wsalloc(ws, off, sizeof(int) * 1024);
  float*  hseq  = (float*) wsalloc(ws, off, sizeof(float) * L * 128);
  half_t* hseqH = (half_t*)wsalloc(ws, off, sizeof(half_t) * L * 128);
  int*    bkt   = (int*)   wsalloc(ws, off, sizeof(int) * (long)L * L);
  float*  qkv   = (float*) wsalloc(ws, off, sizeof(float) * L * 384);
  float*  attnO = (float*) wsalloc(ws, off, sizeof(float) * L * 128);
  half_t* attnOH= (half_t*)wsalloc(ws, off, sizeof(half_t) * L * 128);
  float*  tmp128= (float*) wsalloc(ws, off, sizeof(float) * L * 128);
  float*  f1    = (float*) wsalloc(ws, off, sizeof(float) * (long)L * 2048);
  half_t* f1H   = (half_t*)wsalloc(ws, off, sizeof(half_t) * (long)L * 2048);
  half_t* wipH  = (half_t*)wsalloc(ws, off, sizeof(half_t) * 384 * 128);
  half_t* wopH  = (half_t*)wsalloc(ws, off, sizeof(half_t) * 128 * 128);
  half_t* wf1H  = (half_t*)wsalloc(ws, off, sizeof(half_t) * 2048 * 128);
  half_t* wf2H  = (half_t*)wsalloc(ws, off, sizeof(half_t) * 2048 * 128);
  float*  dec   = (float*) wsalloc(ws, off, sizeof(float) * 1024 * 128);
  half_t* decH  = (half_t*)wsalloc(ws, off, sizeof(half_t) * 1024 * 128);

  // ---------------- encoder convs ----------------
  k_conv3x3<<<NBLK(H1), 256, 0, stream>>>(x_blocks, enc_c1_w, enc_c1_b, bigA,
                                          1024, 2, 50, 50, 32, 50, 50, 1, 1);
  k_conv3x3<<<NBLK(H2), 256, 0, stream>>>(bigA, enc_c2_w, enc_c2_b, bigB,
                                          1024, 32, 50, 50, 64, 25, 25, 2, 1);
  k_conv3x3<<<NBLK(H3), 256, 0, stream>>>(bigB, enc_c3_w, enc_c3_b, bigC,
                                          1024, 64, 25, 25, 64, 13, 13, 2, 1);

  // ---------------- patch projection (WMMA) ----------------
  k_f32_to_f16<<<NBLK(H3), 256, 0, stream>>>(bigC, bigH, H3);
  k_transpose_f16<<<NBLK(10816L * 128), 256, 0, stream>>>(enc_lin_w, wbigH, 10816, 128);
  {
    dim3 g(128 / 64, (1024 + 31) / 32);
    k_wmma_gemm<<<g, 256, 0, stream>>>(bigH, wbigH, enc_lin_b, xemb, 1024, 128, 10816, 0);
  }
  k_pos_add<<<NBLK(1024 * 128), 256, 0, stream>>>(xemb, row_embed, col_embed);

  // ---------------- masking ----------------
  k_argsort<<<1, 256, 0, stream>>>(noise, ids, 1024);
  k_gather<<<NBLK((long)L * 128), 256, 0, stream>>>(xemb, ids, hseq, L);
  k_buckets<<<NBLK((long)L * L), 256, 0, stream>>>(bkt, L);

  // ---------------- transformer ----------------
  for (int i = 0; i < 4; ++i) {
    k_transpose_f16<<<NBLK(128L * 384), 256, 0, stream>>>(in_proj_w + (long)i * 128 * 384, wipH, 128, 384);
    k_transpose_f16<<<NBLK(128L * 128), 256, 0, stream>>>(out_proj_w + (long)i * 128 * 128, wopH, 128, 128);
    k_transpose_f16<<<NBLK(128L * 2048), 256, 0, stream>>>(ff1_w + (long)i * 128 * 2048, wf1H, 128, 2048);
    k_transpose_f16<<<NBLK(2048L * 128), 256, 0, stream>>>(ff2_w + (long)i * 2048 * 128, wf2H, 2048, 128);

    k_f32_to_f16<<<NBLK((long)L * 128), 256, 0, stream>>>(hseq, hseqH, (long)L * 128);
    {
      dim3 g(384 / 64, (L + 31) / 32);
      k_wmma_gemm<<<g, 256, 0, stream>>>(hseqH, wipH, in_proj_b + (long)i * 384, qkv, L, 384, 128, 0);
    }
    {
      dim3 g(L, 8);
      k_attn<<<g, 256, 0, stream>>>(qkv, bkt, rel_tab + (long)i * 32 * 8, attnO, L);
    }
    k_f32_to_f16<<<NBLK((long)L * 128), 256, 0, stream>>>(attnO, attnOH, (long)L * 128);
    {
      dim3 g(128 / 64, (L + 31) / 32);
      k_wmma_gemm<<<g, 256, 0, stream>>>(attnOH, wopH, out_proj_b + (long)i * 128, tmp128, L, 128, 128, 0);
    }
    k_add_ln<<<L, 128, 0, stream>>>(hseq, tmp128, ln1_g + (long)i * 128, ln1_b + (long)i * 128);

    k_f32_to_f16<<<NBLK((long)L * 128), 256, 0, stream>>>(hseq, hseqH, (long)L * 128);
    {
      dim3 g(2048 / 64, (L + 31) / 32);
      k_wmma_gemm<<<g, 256, 0, stream>>>(hseqH, wf1H, ff1_b + (long)i * 2048, f1, L, 2048, 128, 2);
    }
    k_f32_to_f16<<<NBLK((long)L * 2048), 256, 0, stream>>>(f1, f1H, (long)L * 2048);
    {
      dim3 g(128 / 64, (L + 31) / 32);
      k_wmma_gemm<<<g, 256, 0, stream>>>(f1H, wf2H, ff2_b + (long)i * 128, tmp128, L, 128, 2048, 0);
    }
    k_add_ln<<<L, 128, 0, stream>>>(hseq, tmp128, ln2_g + (long)i * 128, ln2_b + (long)i * 128);
  }

  // ---------------- decoder ----------------
  k_zero_f32<<<NBLK(1024 * 128), 256, 0, stream>>>(dec, 1024 * 128);
  k_scatter<<<NBLK((long)L * 128), 256, 0, stream>>>(hseq, ids, dec, L);
  k_f32_to_f16<<<NBLK(1024 * 128), 256, 0, stream>>>(dec, decH, 1024 * 128);
  k_transpose_f16<<<NBLK(128L * 10816), 256, 0, stream>>>(dec_lin_w, wbigH, 128, 10816);
  {
    dim3 g(10816 / 64, (1024 + 31) / 32);
    k_wmma_gemm<<<g, 256, 0, stream>>>(decH, wbigH, dec_lin_b, bigC /*lin*/, 1024, 10816, 128, 1);
  }
  // lin (BN,64,13,13) -> transposed convs
  k_convT3x3<<<NBLK(U1), 256, 0, stream>>>(bigC, dt1_w, dt1_b, bigB, 1024, 64, 13, 13, 64, 1);
  k_convT3x3<<<NBLK(U2), 256, 0, stream>>>(bigB, dt2_w, dt2_b, bigA, 1024, 64, 26, 26, 32, 1);
  k_conv3x3<<<NBLK(PP), 256, 0, stream>>>(bigA, dc_w, dc_b, bigC /*p*/,
                                          1024, 32, 52, 52, 2, 52, 52, 1, 0);

  // ---------------- outputs ----------------
  float* out = (float*)d_out;
  k_out_transpose<<<NBLK(PP), 256, 0, stream>>>(bigC, out);
  k_out_mask<<<2, 256, 0, stream>>>(ids, out + PP);
}
